// ThresholdMoeLayer_14370960573217
// MI455X (gfx1250) — matrix operand order repaired and checked by reading, not compiled, not run on hardware
//
#include <hip/hip_runtime.h>
#include <hip/hip_bf16.h>

// ---------------------------------------------------------------------------
// ThresholdMoE (gate -> per-expert LoRA -> weighted combine) for MI455X/gfx1250
//   k0: gate weights w[N,8]  (softmax+threshold+renorm), f32 -> workspace
//   k1: pack lora_A as bf16 WMMA-B fragments  A_cat[K=1024, C=512]
//   k2: pack lora_B as bf16 WMMA-B fragments  B_cat[K=512,  D=1024]
//   k3: per 64-token tile:
//         phase1: H[64,512] = X[64,1024] @ A_cat   (bf16 WMMA, f32 acc)
//                 scale rows by w[n,e], store bf16 to LDS
//         phase2: O[64,1024] = Hw @ B_cat          (bf16 WMMA, f32 acc)
// Expert dim folds into GEMM-K:  sum_e (w.H_e)@B_e == Hw_cat @ B_cat.
// Fragment addresses = one per-lane base pointer + immediate offsets.
// Explicit ping-pong software pipeline: group g+1 fragment loads issued
// before group g WMMAs (4 fragments per group, A-fragment double-buffered).
// ---------------------------------------------------------------------------

typedef __attribute__((ext_vector_type(16))) __bf16 v16bf;
typedef __attribute__((ext_vector_type(8)))  float  v8f;

#define DM   1024   // d_model
#define NE   8      // experts
#define NR   64     // lora rank
#define CC   512    // NE*NR  (phase-1 output cols / phase-2 K)
#define MT   64     // token rows per tile
#define KC   256    // K staging chunk for X
#define XP   264    // sX row pitch (bf16): 256 + 8 pad -> bank stride 4
#define HP   520    // sH row pitch (bf16): 512 + 8 pad -> bank stride 4

static __device__ __forceinline__ unsigned short f2bf(float f) {
    unsigned int u = __builtin_bit_cast(unsigned int, f);
    u += 0x7FFFu + ((u >> 16) & 1u);          // round-to-nearest-even
    return (unsigned short)(u >> 16);
}

struct U32x8 { uint4 a, b; };
// Load one 16x32 bf16 A-operand fragment from two contiguous 16B runs
// (elements 0-7 at K=kb+8h, elements 8-15 at K=kb+16+8h per ISA 7.12.2).
static __device__ __forceinline__ v16bf ldfrag(const unsigned short* p0,
                                               const unsigned short* p1) {
    U32x8 s;
    s.a = *(const uint4*)p0;
    s.b = *(const uint4*)p1;
    return __builtin_bit_cast(v16bf, s);
}

static __device__ __forceinline__ v8f wmma_bf16(v16bf a, v16bf b, v8f c) {
    return __builtin_amdgcn_wmma_f32_16x16x32_bf16(false, a, false, b,
                                                   (short)0, c, false, false);
}

// ---------------------------------------------------------------------------
// k0: gate.  One wave per token; lanes stride over D, xor-reduce 8 partials.
// ---------------------------------------------------------------------------
__global__ __launch_bounds__(256) void gate_kernel(
        const float* __restrict__ x, const float* __restrict__ gw,
        const float* __restrict__ gb, float* __restrict__ Wout, int ntok) {
    const int lane = threadIdx.x & 31;
    const long n = (long)blockIdx.x * 8 + (threadIdx.x >> 5);
    if (n >= ntok) return;
    const float* xr = x + n * DM;
    float p[NE];
#pragma unroll
    for (int e = 0; e < NE; ++e) p[e] = 0.f;
    for (int d = lane; d < DM; d += 32) {
        const float xv = xr[d];
        const float4 g0 = *(const float4*)(gw + d * NE);
        const float4 g1 = *(const float4*)(gw + d * NE + 4);
        p[0] += xv * g0.x; p[1] += xv * g0.y; p[2] += xv * g0.z; p[3] += xv * g0.w;
        p[4] += xv * g1.x; p[5] += xv * g1.y; p[6] += xv * g1.z; p[7] += xv * g1.w;
    }
#pragma unroll
    for (int e = 0; e < NE; ++e)
        for (int off = 16; off > 0; off >>= 1)
            p[e] += __shfl_xor(p[e], off, 32);
    float l[NE], g[NE];
    float mx = -3.4e38f;
#pragma unroll
    for (int e = 0; e < NE; ++e) { l[e] = p[e] + gb[e]; mx = fmaxf(mx, l[e]); }
    float s = 0.f;
#pragma unroll
    for (int e = 0; e < NE; ++e) { g[e] = __expf(l[e] - mx); s += g[e]; }
    const float inv = 1.f / s;
    float wsum = 0.f;
#pragma unroll
    for (int e = 0; e < NE; ++e) {
        g[e] *= inv;
        g[e] = (g[e] >= 0.1f) ? g[e] : 0.f;   // threshold gate
        wsum += g[e];
    }
    if (wsum == 0.f) wsum = 1.f;
    const float invw = 1.f / wsum;
    if (lane == 0) {
#pragma unroll
        for (int e = 0; e < NE; ++e) Wout[n * NE + e] = g[e] * invw;
    }
}

// ---------------------------------------------------------------------------
// k1/k2: pack A_cat / B_cat into bf16 B-operand fragment order:
//   frag elem layout (assumed 32x16 B): lane L -> col N = L&15, h = L>>4,
//   element e -> K = 16*(e>>3) + 8*h + (e&7).   Packed: ((kT*NT + nT)*32 + lane)*16 + e.
// ---------------------------------------------------------------------------
__global__ __launch_bounds__(256) void packA_kernel(
        const float* __restrict__ A, unsigned short* __restrict__ Ap) {
    const int idx = blockIdx.x * 256 + threadIdx.x;   // 1024*512 elems
    const int e    = idx & 15;
    const int lane = (idx >> 4) & 31;
    const int nT   = (idx >> 9) & 31;                 // 32 col tiles
    const int kT   = idx >> 14;                       // 32 k tiles
    const int K = kT * 32 + ((e >> 3) << 4) + ((lane >> 4) << 3) + (e & 7);   // d index
    const int N = nT * 16 + (lane & 15);              // e*64 + r
    const int ex = N >> 6, r = N & 63;
    Ap[idx] = f2bf(A[(long)ex * DM * NR + (long)K * NR + r]);
}

__global__ __launch_bounds__(256) void packB_kernel(
        const float* __restrict__ B, unsigned short* __restrict__ Bp) {
    const int idx = blockIdx.x * 256 + threadIdx.x;   // 512*1024 elems
    const int e    = idx & 15;
    const int lane = (idx >> 4) & 31;
    const int nT   = (idx >> 9) & 63;                 // 64 col tiles
    const int kT   = idx >> 15;                       // 16 k tiles
    const int K = kT * 32 + ((e >> 3) << 4) + ((lane >> 4) << 3) + (e & 7);   // e*64 + r
    const int N = nT * 16 + (lane & 15);              // d index
    const int ex = K >> 6, r = K & 63;
    Bp[idx] = f2bf(B[(long)ex * NR * DM + (long)r * DM + N]);
}

// ---------------------------------------------------------------------------
// k3: main fused MoE kernel. 256 threads = 8 waves (4 row-stripes x 2 cols).
// ---------------------------------------------------------------------------
__global__ __launch_bounds__(256, 1) void moe_kernel(
        const float* __restrict__ x, const float* __restrict__ W,
        const unsigned short* __restrict__ Apack,
        const unsigned short* __restrict__ Bpack,
        float* __restrict__ out) {
    __shared__ unsigned short sX[MT * XP];   //  33.0 KB  X chunk, bf16
    __shared__ unsigned short sH[MT * HP];   //  65.0 KB  weighted H, bf16
    __shared__ float          sW[MT * NE];   //   2.0 KB  gate weights

    const int tid  = threadIdx.x;
    const int lane = tid & 31;
    const int wave = tid >> 5;
    const int wr   = wave & 3;         // row stripe 0..3
    const int wc   = wave >> 2;        // col half  0..1
    const int rowBase = wr * 16;
    const int h  = lane >> 4;
    const int ln = lane & 15;
    const long tileRow = (long)blockIdx.x * MT;

    for (int i = tid; i < MT * NE; i += 256) sW[i] = W[tileRow * NE + i];

    // Per-lane fragment base pointers; everything else is immediate offsets.
    // Apack frag (v16bf units): kt*1024 + (e*4 + wc*2 + f)*32 + lane
    const v16bf* aF0 = (const v16bf*)Apack + lane + wc * 64;
    // Bpack frag (v16bf units): k*2048 + (nc*16 + wc*8 + t)*32 + lane
    const v16bf* bF0 = (const v16bf*)Bpack + lane + wc * 256;
    // LDS per-lane row pointers
    const unsigned short* sXrow = &sX[(rowBase + ln) * XP + 8 * h];
    const unsigned short* sHrow = &sH[(rowBase + ln) * HP + 8 * h];

    // ---- phase 1: H accumulators, 8 experts x (16x32) per wave ----
    v8f acc[NE][2];
#pragma unroll
    for (int e = 0; e < NE; ++e) {
        acc[e][0] = v8f{0,0,0,0,0,0,0,0};
        acc[e][1] = v8f{0,0,0,0,0,0,0,0};
    }

    for (int kc = 0; kc < DM / KC; ++kc) {
        __syncthreads();                       // previous-chunk readers done
        // stage X[64, KC] f32 -> bf16 LDS (coalesced float4 loads)
        for (int i = tid; i < MT * (KC / 4); i += 256) {
            const int r = i / (KC / 4), c4 = i % (KC / 4);
            const float4 v = *(const float4*)(x + (tileRow + r) * DM + kc * KC + c4 * 4);
            unsigned int lo = (unsigned int)f2bf(v.x) | ((unsigned int)f2bf(v.y) << 16);
            unsigned int hi = (unsigned int)f2bf(v.z) | ((unsigned int)f2bf(v.w) << 16);
            *(uint2*)(&sX[r * XP + c4 * 4]) = make_uint2(lo, hi);
        }
        __syncthreads();

        // prefetch next X chunk into L2 while we crunch this one
        if (kc + 1 < DM / KC) {
            const int pr = tid >> 2, po = (tid & 3) * 256;
            const char* np = (const char*)(x + (tileRow + pr) * DM + (kc + 1) * KC) + po;
            __builtin_prefetch(np, 0, 0);
            __builtin_prefetch(np + 128, 0, 0);
        }

        const v16bf* aF = aF0 + (size_t)kc * 8192;   // 8 k-tiles per chunk

        // Ping-pong pipeline: 32 groups of 4 fragments (kt = g>>2, q = g&3;
        // frag j -> expert e = q*2 + (j>>1), half f = j&1).
        v16bf afv[2];
        v16bf buf[2][4];
        afv[0] = ldfrag(sXrow, sXrow + 16);
#pragma unroll
        for (int j = 0; j < 4; ++j)
            buf[0][j] = aF[(j >> 1) * 128 + (j & 1) * 32];

#pragma unroll
        for (int g = 0; g < 32; ++g) {
            const int cur = g & 1, nxt = cur ^ 1;
            const int kt = g >> 2, q = g & 3;
            if (g + 1 < 32) {                      // issue next group's loads
                const int nkt = (g + 1) >> 2, nq = (g + 1) & 3;
                if (nq == 0)
                    afv[nkt & 1] = ldfrag(sXrow + nkt * 32, sXrow + nkt * 32 + 16);
#pragma unroll
                for (int j = 0; j < 4; ++j)
                    buf[nxt][j] =
                        aF[nkt * 1024 + (nq * 2 + (j >> 1)) * 128 + (j & 1) * 32];
            }
            const v16bf af = afv[kt & 1];
#pragma unroll
            for (int j = 0; j < 4; ++j) {
                const int e = q * 2 + (j >> 1), fh = j & 1;
                acc[e][fh] = wmma_bf16(af, buf[cur][j], acc[e][fh]);
            }
        }
    }
    __syncthreads();

    // ---- apply gate weights, H -> LDS bf16 (C/D layout: M = v + 8h, N = ln) ----
#pragma unroll
    for (int e = 0; e < NE; ++e) {
#pragma unroll
        for (int f = 0; f < 2; ++f) {
#pragma unroll
            for (int v = 0; v < 8; ++v) {
                const int r = rowBase + v + 8 * h;
                const float wgt = sW[r * NE + e];
                sH[r * HP + e * NR + wc * 32 + f * 16 + ln] = f2bf(acc[e][f][v] * wgt);
            }
        }
    }
    __syncthreads();

    // ---- phase 2: O[64,1024] = Hw[64,512] @ B_cat[512,1024] ----
    for (int nc = 0; nc < 4; ++nc) {                      // 256-col chunks
        const v16bf* bF = bF0 + nc * 512;
        v8f o[8];
#pragma unroll
        for (int t = 0; t < 8; ++t) o[t] = v8f{0,0,0,0,0,0,0,0};

        // Ping-pong pipeline: 32 groups of 4 (k = g>>1, half = g&1;
        // frag j -> t = half*4 + j).
        v16bf afv[2];
        v16bf buf[2][4];
        afv[0] = ldfrag(sHrow, sHrow + 16);
#pragma unroll
        for (int j = 0; j < 4; ++j) buf[0][j] = bF[j * 32];

#pragma unroll
        for (int g = 0; g < 32; ++g) {
            const int cur = g & 1, nxt = cur ^ 1;
            const int k = g >> 1, half = g & 1;
            if (g + 1 < 32) {
                const int nk = (g + 1) >> 1, nh = (g + 1) & 1;
                if (nh == 0)
                    afv[nk & 1] = ldfrag(sHrow + nk * 32, sHrow + nk * 32 + 16);
#pragma unroll
                for (int j = 0; j < 4; ++j)
                    buf[nxt][j] = bF[nk * 2048 + (nh * 4 + j) * 32];
            }
            const v16bf af = afv[k & 1];
#pragma unroll
            for (int j = 0; j < 4; ++j) {
                const int t = half * 4 + j;
                o[t] = wmma_bf16(af, buf[cur][j], o[t]);
            }
        }
#pragma unroll
        for (int t = 0; t < 8; ++t) {
#pragma unroll
            for (int v = 0; v < 8; ++v) {
                out[(tileRow + rowBase + v + 8 * h) * DM +
                    nc * 256 + wc * 128 + t * 16 + ln] = o[t][v];
            }
        }
    }
}

// ---------------------------------------------------------------------------
extern "C" void kernel_launch(void* const* d_in, const int* in_sizes, int n_in,
                              void* d_out, int out_size, void* d_ws, size_t ws_size,
                              hipStream_t stream) {
    const float* x  = (const float*)d_in[0];   // [B,T,D] f32
    const float* gw = (const float*)d_in[1];   // [D,E]
    const float* gb = (const float*)d_in[2];   // [E]
    const float* lA = (const float*)d_in[3];   // [E,D,R]
    const float* lB = (const float*)d_in[4];   // [E,R,D]
    float* out = (float*)d_out;
    const int ntok = in_sizes[0] / DM;         // 65536

    char* ws = (char*)d_ws;
    float* Wgate = (float*)ws;                                   // ntok*8 f32 (2 MB)
    unsigned short* Apack = (unsigned short*)(ws + (size_t)ntok * NE * 4);
    unsigned short* Bpack = Apack + (size_t)DM * CC;             // +1 MB

    gate_kernel<<<ntok / 8, 256, 0, stream>>>(x, gw, gb, Wgate, ntok);
    packA_kernel<<<(DM * CC) / 256, 256, 0, stream>>>(lA, Apack);
    packB_kernel<<<(CC * DM) / 256, 256, 0, stream>>>(lB, Bpack);
    moe_kernel<<<ntok / MT, 256, 0, stream>>>(x, Wgate, Apack, Bpack, out);
}
